// MultiHeadAttention_13657996001983
// MI455X (gfx1250) — matrix operand order repaired and checked by reading, not compiled, not run on hardware
//
#include <hip/hip_runtime.h>
#include <hip/hip_bf16.h>

// ---------------------------------------------------------------------------
// Sliding-window causal MHA for MI455X (gfx1250, wave32, WMMA).
// Pipeline: f32->f16 convert -> WMMA GEMM (qkv) -> WMMA flash attention
//           (window=128) -> WMMA GEMM (out proj, f32 result).
// GEMM stages tiles with GLOBAL_LOAD_ASYNC_TO_LDS_B128 (ASYNCcnt) into
// double-buffered LDS; the loop is manually 2x-unrolled so buffer indices
// are compile-time constants and the pipeline is branch-light.
// ---------------------------------------------------------------------------

typedef __attribute__((ext_vector_type(16))) _Float16 v16h;
typedef __attribute__((ext_vector_type(4)))  _Float16 v4h;
typedef __attribute__((ext_vector_type(8)))  float    v8f;

union F16x16 { v16h v; uint4 q[2]; };

#define EMB   1024
#define T_SEQ 2048
#define NH    16
#define HD    64
#define WIN   128
#define QKV_LD 3072
#define BATCH 4

// ---- gfx1250 async global->LDS copy (16B per lane), tracked by ASYNCcnt ----
__device__ __forceinline__ unsigned lds_off32(const void* p) {
  // generic pointer to a __shared__ object: addr[31:0] is the LDS offset
  return (unsigned)(unsigned long long)p;
}
__device__ __forceinline__ void async_copy_b128(unsigned lds_addr, const void* gptr) {
  unsigned long long ga = (unsigned long long)gptr;
  asm volatile("global_load_async_to_lds_b128 %0, %1, off"
               :: "v"(lds_addr), "v"(ga) : "memory");
}
#define S_WAIT_ASYNCCNT(n) asm volatile("s_wait_asynccnt " #n ::: "memory")

// ---------------------------------------------------------------------------
// f32 -> f16 convert, 4 elements / thread (all sizes are multiples of 1024)
// ---------------------------------------------------------------------------
__global__ __launch_bounds__(256)
void cvt_f32_to_f16(const float* __restrict__ src, _Float16* __restrict__ dst, int n) {
  int i = (blockIdx.x * 256 + threadIdx.x) * 4;
  if (i < n) {
    float4 f = *(const float4*)(src + i);
    v4h h = {(_Float16)f.x, (_Float16)f.y, (_Float16)f.z, (_Float16)f.w};
    *(v4h*)(dst + i) = h;
  }
}

// ---------------------------------------------------------------------------
// C[M,N] = A[M,K] * W[N,K]^T   (both f16 row-major over K; f32 accumulate)
// Block tile 128x128, 8 waves (4 M x 2 N), wave tile 32x64 = 2x4 wmma tiles.
// K staged through double-buffered LDS in chunks of 32 with +8 half padding
// per row (row stride 80B = 20 dwords -> conflict-free 16-lane frag reads).
// Requires K % 64 == 0 (both calls use K = 1024).
// ---------------------------------------------------------------------------
#define TILE_K     32
#define LDS_STRIDE 40   // halfs per row: 32 + 8 pad

__device__ __forceinline__ void gemm_stage(
    _Float16* __restrict__ bufA, _Float16* __restrict__ bufB,
    const _Float16* __restrict__ A, const _Float16* __restrict__ W,
    int mb, int nb, int K, int k0,
    int row0, int seg0, int row1, int seg1) {
  async_copy_b128(lds_off32(bufA + row0 * LDS_STRIDE + seg0 * 8),
                  A + (size_t)(mb + row0) * K + k0 + seg0 * 8);
  async_copy_b128(lds_off32(bufB + row0 * LDS_STRIDE + seg0 * 8),
                  W + (size_t)(nb + row0) * K + k0 + seg0 * 8);
  async_copy_b128(lds_off32(bufA + row1 * LDS_STRIDE + seg1 * 8),
                  A + (size_t)(mb + row1) * K + k0 + seg1 * 8);
  async_copy_b128(lds_off32(bufB + row1 * LDS_STRIDE + seg1 * 8),
                  W + (size_t)(nb + row1) * K + k0 + seg1 * 8);
}

__device__ __forceinline__ void gemm_tile_compute(
    const _Float16* __restrict__ bufA, const _Float16* __restrict__ bufB,
    int wm, int wn, int l16, int hb, v8f acc[2][4]) {
  F16x16 aF[2], bF[4];
#pragma unroll
  for (int im = 0; im < 2; ++im) {
    const _Float16* base = bufA + (wm * 32 + im * 16 + l16) * LDS_STRIDE;
    // A 16x32 layout: lane<16 -> k {0..7,16..23}; lane>=16 -> k {8..15,24..31}
    aF[im].q[0] = *(const uint4*)(base + hb * 8);
    aF[im].q[1] = *(const uint4*)(base + 16 + hb * 8);
  }
#pragma unroll
  for (int in = 0; in < 4; ++in) {
    const _Float16* base = bufB + (wn * 64 + in * 16 + l16) * LDS_STRIDE + hb * 8;
    // B 32x16 layout: lane<16 -> k {0..7,16..23}; lane>=16 -> k {8..15,24..31} (lane = N col)
    bF[in].q[0] = *(const uint4*)(base);
    bF[in].q[1] = *(const uint4*)(base + 16);
  }
#pragma unroll
  for (int im = 0; im < 2; ++im)
#pragma unroll
    for (int in = 0; in < 4; ++in)
      acc[im][in] = __builtin_amdgcn_wmma_f32_16x16x32_f16(
          false, aF[im].v, false, bF[in].v, (short)0, acc[im][in], false, false);
}

template <bool F32OUT>
__global__ __launch_bounds__(256)
void wmma_gemm_nt(const _Float16* __restrict__ A, const _Float16* __restrict__ W,
                  void* __restrict__ outp, int M, int N, int K) {
  __shared__ _Float16 lA[2][128 * LDS_STRIDE];
  __shared__ _Float16 lB[2][128 * LDS_STRIDE];

  const int tid  = threadIdx.x;
  const int lane = tid & 31;
  const int wv   = tid >> 5;
  const int l16  = lane & 15;
  const int hb   = lane >> 4;          // which 16-lane half of the wave
  const int mb   = blockIdx.y * 128;
  const int nb   = blockIdx.x * 128;
  const int wm   = wv & 3;             // wave row (0..3)  -> 32 M rows
  const int wn   = wv >> 2;            // wave col (0..1)  -> 64 N cols

  // this thread's two staging slots: row = slot/4, 16B segment = slot%4
  const int row0 = tid >> 2,         seg0 = tid & 3;
  const int row1 = (tid + 256) >> 2, seg1 = tid & 3;

  v8f acc[2][4];
#pragma unroll
  for (int im = 0; im < 2; ++im)
#pragma unroll
    for (int in = 0; in < 4; ++in) acc[im][in] = {};

  const int nK = K / TILE_K;           // even (K % 64 == 0)

  // prologue: tile 0 -> buffer 0
  gemm_stage(lA[0], lB[0], A, W, mb, nb, K, 0, row0, seg0, row1, seg1);

  for (int kt = 0; kt < nK; kt += 2) {
    // ---- sub-iter A: compute buf0 (tile kt), stream tile kt+1 -> buf1 ----
    gemm_stage(lA[1], lB[1], A, W, mb, nb, K, (kt + 1) * TILE_K,
               row0, seg0, row1, seg1);
    S_WAIT_ASYNCCNT(0x4);              // async loads retire in order
    __syncthreads();
    gemm_tile_compute(lA[0], lB[0], wm, wn, l16, hb, acc);
    __syncthreads();                   // buf0 reads consumed

    // ---- sub-iter B: compute buf1 (tile kt+1), stream tile kt+2 -> buf0 ----
    if (kt + 2 < nK) {
      gemm_stage(lA[0], lB[0], A, W, mb, nb, K, (kt + 2) * TILE_K,
                 row0, seg0, row1, seg1);
      S_WAIT_ASYNCCNT(0x4);
    } else {
      S_WAIT_ASYNCCNT(0x0);
    }
    __syncthreads();
    gemm_tile_compute(lA[1], lB[1], wm, wn, l16, hb, acc);
    __syncthreads();                   // buf1 reads consumed
  }

  // ---- epilogue: C layout VGPR r -> row r + 8*hb, col = l16 ----
#pragma unroll
  for (int im = 0; im < 2; ++im) {
    const int rowbase = mb + wm * 32 + im * 16 + hb * 8;
#pragma unroll
    for (int in = 0; in < 4; ++in) {
      const int col = nb + wn * 64 + in * 16 + l16;
#pragma unroll
      for (int r = 0; r < 8; ++r) {
        float val = acc[im][in][r];
        if (F32OUT) ((float*)outp)[(size_t)(rowbase + r) * N + col] = val;
        else        ((_Float16*)outp)[(size_t)(rowbase + r) * N + col] = (_Float16)val;
      }
    }
  }
}

// ---------------------------------------------------------------------------
// Windowed causal flash attention. One wave per (b, h, 16-query tile).
// Computes S^T = K * Q^T with WMMA (both operands contiguous over d);
// the C-layout of two consecutive S^T tiles is exactly the 16x32 f16
// A-operand layout for P*V — no cross-lane shuffles to feed the PV wmma.
// qkv layout: [B, T, 3, H, D] f16 (stride QKV_LD per t).
// ---------------------------------------------------------------------------
__global__ __launch_bounds__(256)
void attn_swa(const _Float16* __restrict__ qkv, _Float16* __restrict__ attn_out) {
  const int tid  = threadIdx.x;
  const int lane = tid & 31;
  const int l16  = lane & 15;
  const int hb   = lane >> 4;
  const int wid  = (blockIdx.x * 256 + tid) >> 5;
  const int nqt  = T_SEQ / 16;                 // 128
  const int qt   = wid % nqt;
  const int h    = (wid / nqt) % NH;
  const int b    = wid / (nqt * NH);
  const int i0   = qt * 16;

  const size_t rowB = (size_t)b * T_SEQ;
  const _Float16* qbase = qkv + rowB * QKV_LD + h * HD;
  const _Float16* kbase = qbase + EMB;         // +1024
  const _Float16* vbase = qbase + 2 * EMB;     // +2048

  // Q as B operand: lane = query column; lane<16 -> d {0..7,16..23}, lane>=16 -> d {8..15,24..31}
  F16x16 qb[2];
  {
    const _Float16* p = qbase + (size_t)(i0 + l16) * QKV_LD + hb * 8;
    qb[0].q[0] = *(const uint4*)(p);
    qb[0].q[1] = *(const uint4*)(p + 16);
    qb[1].q[0] = *(const uint4*)(p + 32);
    qb[1].q[1] = *(const uint4*)(p + 48);
  }

  float m_run = -1e30f, l_run = 0.0f;
  v8f o[4];
#pragma unroll
  for (int t = 0; t < 4; ++t) o[t] = {};

  const int jstart = (i0 >= WIN) ? (i0 - WIN) : 0;
  const int nkt    = (i0 + 16 - jstart) >> 4;   // 1..9 key tiles of 16
  const int npair  = (nkt + 1) >> 1;
  const float scale = 0.125f;                   // 1/sqrt(64)
  const int i_q = i0 + l16;                     // this lane-pair's query row

  for (int pidx = 0; pidx < npair; ++pidx) {
    const int j0 = jstart + pidx * 32;

    // ---- S^T tiles (key-major) via wmma: st[t][r] = score(key j0+16t+r+8*hb, query i_q)
    v8f st[2];
#pragma unroll
    for (int t = 0; t < 2; ++t) {
      int krow = j0 + 16 * t + l16;
      if (krow > T_SEQ - 1) krow = T_SEQ - 1;   // clamp; masked below anyway
      const _Float16* p = kbase + (size_t)krow * QKV_LD + hb * 8;
      F16x16 ka0, ka1;
      ka0.q[0] = *(const uint4*)(p);
      ka0.q[1] = *(const uint4*)(p + 16);
      ka1.q[0] = *(const uint4*)(p + 32);
      ka1.q[1] = *(const uint4*)(p + 48);
      v8f c = {};
      c = __builtin_amdgcn_wmma_f32_16x16x32_f16(false, ka0.v, false, qb[0].v,
                                                 (short)0, c, false, false);
      c = __builtin_amdgcn_wmma_f32_16x16x32_f16(false, ka1.v, false, qb[1].v,
                                                 (short)0, c, false, false);
      st[t] = c;
    }

    // ---- mask + scale + tile max (per query = per lane pair) ----
    float tmax = -__builtin_inff();
#pragma unroll
    for (int t = 0; t < 2; ++t)
#pragma unroll
      for (int r = 0; r < 8; ++r) {
        int j = j0 + 16 * t + r + 8 * hb;
        bool ok = (j <= i_q) && (j >= i_q - WIN);
        float s = ok ? st[t][r] * scale : -__builtin_inff();
        st[t][r] = s;
        tmax = fmaxf(tmax, s);
      }
    tmax = fmaxf(tmax, __shfl_xor(tmax, 16, 32));

    const float m_new = fmaxf(m_run, tmax);
    const float fr = __expf(m_run - m_new);

    // ---- probabilities; S^T C-layout == PV A-operand layout directly ----
    float lsum = 0.0f;
    F16x16 pa;
#pragma unroll
    for (int t = 0; t < 2; ++t)
#pragma unroll
      for (int r = 0; r < 8; ++r) {
        float p = __expf(st[t][r] - m_new);
        lsum += p;
        pa.v[t * 8 + r] = (_Float16)p;
      }
    lsum += __shfl_xor(lsum, 16, 32);
    l_run = l_run * fr + lsum;
    m_run = m_new;

    // ---- rescale O accumulators: lane's O rows are queries r + 8*hb ----
    float fq[8];
#pragma unroll
    for (int r = 0; r < 8; ++r) fq[r] = __shfl(fr, r + hb * 8, 32);
#pragma unroll
    for (int t = 0; t < 4; ++t)
#pragma unroll
      for (int r = 0; r < 8; ++r) o[t][r] *= fq[r];

    // ---- P * V : V as B operand (lane = d column, halfs = key index) ----
#pragma unroll
    for (int t = 0; t < 4; ++t) {
      F16x16 vb;
#pragma unroll
      for (int kk = 0; kk < 16; ++kk) {
        int krow = j0 + (kk & 7) + hb * 8 + (kk >> 3) * 16;
        if (krow > T_SEQ - 1) krow = T_SEQ - 1;
        vb.v[kk] = vbase[(size_t)krow * QKV_LD + t * 16 + l16];
      }
      o[t] = __builtin_amdgcn_wmma_f32_16x16x32_f16(false, pa.v, false, vb.v,
                                                    (short)0, o[t], false, false);
    }
  }

  // ---- epilogue: normalize and store f16 [B,T,H*D] ----
  const float inv = 1.0f / l_run;
  float invq[8];
#pragma unroll
  for (int r = 0; r < 8; ++r) invq[r] = __shfl(inv, r + hb * 8, 32);
  _Float16* obase = attn_out + (rowB + i0) * EMB + h * HD;
#pragma unroll
  for (int t = 0; t < 4; ++t)
#pragma unroll
    for (int r = 0; r < 8; ++r)
      obase[(size_t)(r + hb * 8) * EMB + t * 16 + l16] =
          (_Float16)(o[t][r] * invq[r]);
}

// ---------------------------------------------------------------------------
// Host launch
// ---------------------------------------------------------------------------
extern "C" void kernel_launch(void* const* d_in, const int* in_sizes, int n_in,
                              void* d_out, int out_size, void* d_ws, size_t ws_size,
                              hipStream_t stream) {
  const float* x     = (const float*)d_in[0];   // [4,2048,1024]
  const float* w_qkv = (const float*)d_in[1];   // [3072,1024]
  const float* w_out = (const float*)d_in[2];   // [1024,1024]
  float* out = (float*)d_out;                   // [4,2048,1024]

  char* ws = (char*)d_ws;
  const size_t nX = (size_t)BATCH * T_SEQ * EMB;        // 8388608
  const size_t nWq = (size_t)3 * EMB * EMB;             // 3145728
  const size_t nWo = (size_t)EMB * EMB;                 // 1048576
  _Float16* xh    = (_Float16*)(ws);
  _Float16* wqkvh = (_Float16*)(ws + nX * 2);
  _Float16* wouth = (_Float16*)(ws + (nX + nWq) * 2);
  _Float16* qkvh  = (_Float16*)(ws + (nX + nWq + nWo) * 2);
  _Float16* attnh = (_Float16*)(ws + (nX + nWq + nWo + (size_t)BATCH * T_SEQ * QKV_LD) * 2);

  cvt_f32_to_f16<<<(int)(nX  / 1024), 256, 0, stream>>>(x,     xh,    (int)nX);
  cvt_f32_to_f16<<<(int)(nWq / 1024), 256, 0, stream>>>(w_qkv, wqkvh, (int)nWq);
  cvt_f32_to_f16<<<(int)(nWo / 1024), 256, 0, stream>>>(w_out, wouth, (int)nWo);

  const int M = BATCH * T_SEQ;                          // 8192
  dim3 g1(QKV_LD / 128, M / 128);                       // (24, 64)
  wmma_gemm_nt<false><<<g1, 256, 0, stream>>>(xh, wqkvh, (void*)qkvh, M, QKV_LD, EMB);

  const int nwaves = BATCH * NH * (T_SEQ / 16);         // 8192
  attn_swa<<<nwaves / 8, 256, 0, stream>>>(qkvh, attnh);

  dim3 g2(EMB / 128, M / 128);                          // (8, 64)
  wmma_gemm_nt<true><<<g2, 256, 0, stream>>>(attnh, wouth, (void*)out, M, EMB, EMB);
}
